// MyGNNLayer_21303037788728
// MI455X (gfx1250) — compile-verified, matrix operand order checked by reading, not run on hardware
//
#include <hip/hip_runtime.h>
#include <hip/hip_bf16.h>
#include <math.h>

typedef __bf16 bf16;
typedef __attribute__((ext_vector_type(16))) __bf16 v16bf;
typedef __attribute__((ext_vector_type(8)))  __bf16 v8bf;
typedef __attribute__((ext_vector_type(8)))  float  v8f;

#define FDIM   64
#define NNODES 50000
#define NEDGES 500000
#define NGRAPH 64
#define DE  256
#define DN1 320
#define DN2 384
#define DG  448
#define LN_EPS 1e-5f

// workspace layout (bytes)
#define WS_WE    0u            // 2*256*256 bf16 = 262144 B
#define WS_WN1   262144u       // 2*320*320 bf16 = 409600 B
#define WS_WN2   671744u       // 2*384*384 bf16 = 589824 B
#define WS_WG    1261568u      // 2*448*448 bf16 = 802816 B
#define WS_AGG   2097152u      // N*320*4 = 64000000 B
#define WS_CNT   66097152u     // N*4
#define WS_GSUM  66297152u     // B*384*4
#define WS_GCNT  66395456u     // B*4 (padded)
#define WS_ZERO_BYTES (66395712u - WS_AGG)   // multiple of 16

__device__ __forceinline__ float gelu_f(float x) {
    return 0.5f * x * (1.0f + erff(x * 0.7071067811865476f));
}

__device__ __forceinline__ void atomAddF(float* p, float v) {
    __hip_atomic_fetch_add(p, v, __ATOMIC_RELAXED, __HIP_MEMORY_SCOPE_AGENT);
}

// ---------------- weight prep: fp32 [L,D,D] -> bf16 swizzled B-fragments ----
// layout: out[((nb*KB + kb)*32 + lane)*16 + t], lane g=lane>>4, n=nb*16+(lane&15)
//         k = kb*32 + (t&7) + (t>=8 ? 16 : 0) + g*8
__global__ __launch_bounds__(256) void prep_weights(const float* __restrict__ W,
                                                    bf16* __restrict__ out, int D) {
    int idx = blockIdx.x * 256 + threadIdx.x;
    int dd = D * D;
    if (idx >= 2 * dd) return;
    int KB = D / 32;
    int l    = idx / dd;
    int rem  = idx - l * dd;
    int t    = rem & 15;
    int lane = (rem >> 4) & 31;
    int blk  = rem >> 9;
    int kb   = blk % KB;
    int nb   = blk / KB;
    int g    = lane >> 4;
    int n    = nb * 16 + (lane & 15);
    int k    = kb * 32 + (t & 7) + ((t >> 3) << 4) + (g << 3);
    out[idx] = (bf16)W[l * dd + k * D + n];
}

// ---------------- zero workspace accumulators ------------------------------
__global__ __launch_bounds__(256) void zero_ws(float4* __restrict__ p, unsigned n4) {
    float4 z = {0.f, 0.f, 0.f, 0.f};
    for (unsigned i = blockIdx.x * 256u + threadIdx.x; i < n4; i += gridDim.x * 256u)
        p[i] = z;
}

// ---------------- segment counts -------------------------------------------
__global__ __launch_bounds__(256) void counts_kernel(const int* __restrict__ erow,
                                                     const int* __restrict__ batch,
                                                     float* __restrict__ cnt,
                                                     float* __restrict__ gcnt) {
    int i = blockIdx.x * 256 + threadIdx.x;
    if (i < NEDGES)                    atomAddF(&cnt[erow[i]], 1.0f);
    else if (i < NEDGES + NNODES)      atomAddF(&gcnt[batch[i - NEDGES]], 1.0f);
}

// ---------------- one MLP layer over a 64-row tile via WMMA -----------------
// A: LDS bf16 [64][apad] row-major (K = D).  Wsw: swizzled bf16 weights.
// MODE 0: H[m][n] = bf16(gelu(acc+bias))
// MODE 1: Y[m][n] = gelu(acc+bias) + float(Init[m][n])
template <int D, int MODE>
__device__ __forceinline__ void mlp_layer(const bf16* __restrict__ A, int apad,
                                          const bf16* __restrict__ Wsw,
                                          const float* __restrict__ bias,
                                          bf16* __restrict__ H, int hpad,
                                          float* __restrict__ Y, int ypad,
                                          const bf16* __restrict__ Init, int ipad,
                                          int tid) {
    const int KB = D / 32;
    const int NB = D / 16;
    int wave = tid >> 5, lane = tid & 31;
    int g = lane >> 4, ln = lane & 15;
    for (int nb = wave; nb < NB; nb += 8) {
        float bv = bias[nb * 16 + ln];
        v8f acc[4];
        v8f zero = {0.f, 0.f, 0.f, 0.f, 0.f, 0.f, 0.f, 0.f};
#pragma unroll
        for (int mb = 0; mb < 4; ++mb) acc[mb] = zero;
        for (int kb = 0; kb < KB; ++kb) {
            const bf16* bp = Wsw + (((size_t)nb * KB + kb) * 32 + lane) * 16;
            union { v16bf v; v8bf h[2]; } B;
            B.h[0] = *(const v8bf*)(bp);
            B.h[1] = *(const v8bf*)(bp + 8);
            int k0 = kb * 32 + g * 8;
#pragma unroll
            for (int mb = 0; mb < 4; ++mb) {
                const bf16* ap = A + (mb * 16 + ln) * apad + k0;
                union { v16bf v; v8bf h[2]; } Af;
                Af.h[0] = *(const v8bf*)(ap);
                Af.h[1] = *(const v8bf*)(ap + 16);
                acc[mb] = __builtin_amdgcn_wmma_f32_16x16x32_bf16(
                    false, Af.v, false, B.v, (short)0, acc[mb], false, false);
            }
        }
#pragma unroll
        for (int mb = 0; mb < 4; ++mb) {
#pragma unroll
            for (int r = 0; r < 8; ++r) {
                int m = mb * 16 + g * 8 + r;
                int n = nb * 16 + ln;
                float v = gelu_f(acc[mb][r] + bv);
                if (MODE == 0) {
                    H[m * hpad + n] = (bf16)v;
                } else {
                    Y[m * ypad + n] = v + (float)Init[m * ipad + n];
                }
            }
        }
    }
}

// ---------------- parallel LayerNorm stats over 64 rows ---------------------
// 256 threads: 4 partial accumulators per row, combined by threads 0..63.
// Includes internal barriers; callers must barrier before calling.
template <int D>
__device__ __forceinline__ void ln_stats(const float* __restrict__ Y, int ypad,
                                         float* __restrict__ sMu, float* __restrict__ sRs,
                                         float* __restrict__ sPart, int tid) {
    int r = tid >> 2;
    int q = tid & 3;
    float s = 0.f, s2 = 0.f;
    for (int c = q; c < D; c += 4) {
        float v = Y[r * ypad + c];
        s += v; s2 += v * v;
    }
    sPart[tid * 2]     = s;
    sPart[tid * 2 + 1] = s2;
    __syncthreads();
    if (tid < 64) {
        float ss = 0.f, ss2 = 0.f;
#pragma unroll
        for (int q2 = 0; q2 < 4; ++q2) {
            ss  += sPart[(tid * 4 + q2) * 2];
            ss2 += sPart[(tid * 4 + q2) * 2 + 1];
        }
        float m = ss / D;
        sMu[tid] = m;
        sRs[tid] = rsqrtf(fmaxf(ss2 / D - m * m, 0.f) + LN_EPS);
    }
    __syncthreads();
}

// ---------------- fused edge + message kernel ------------------------------
__global__ __launch_bounds__(256) void edge_kernel(
    const float* __restrict__ x, const int* __restrict__ erow,
    const int* __restrict__ ecol, const float* __restrict__ ea,
    const float* __restrict__ u, const int* __restrict__ batch,
    const bf16* __restrict__ wE, const float* __restrict__ eB,
    const float* __restrict__ eG, const float* __restrict__ eBt,
    const bf16* __restrict__ wN1, const float* __restrict__ n1B,
    const float* __restrict__ n1G, const float* __restrict__ n1Bt,
    float* __restrict__ e_out, float* __restrict__ agg) {
    __shared__ bf16  bufA[64 * 336];
    __shared__ bf16  bufB[64 * 336];
    __shared__ float bufY[64 * 328];
    __shared__ int   sRow[64], sCol[64], sB[64];
    __shared__ float sMu[64], sRs[64];
    __shared__ float sPart[512];
    int tid = threadIdx.x;
    long base = (long)blockIdx.x * 64;

    if (tid < 64) {
        long e = base + tid; if (e >= NEDGES) e = NEDGES - 1;
        int r = erow[e], c = ecol[e];
        sRow[tid] = r; sCol[tid] = c; sB[tid] = batch[r];
    }
    __syncthreads();
    // gather e_in = cat[x[row], x[col], edge_attr, u[batch[row]]]  (bf16)
    for (int idx = tid; idx < 64 * DE; idx += 256) {
        int r = idx >> 8, c = idx & 255;
        long e = base + r; if (e >= NEDGES) e = NEDGES - 1;
        float v;
        if (c < 64)        v = x[(long)sRow[r] * 64 + c];
        else if (c < 128)  v = x[(long)sCol[r] * 64 + (c - 64)];
        else if (c < 192)  v = ea[e * 64 + (c - 128)];
        else               v = u[sB[r] * 64 + (c - 192)];
        bufA[r * 336 + c] = (bf16)v;
    }
    __syncthreads();
    mlp_layer<DE, 0>(bufA, 336, wE, eB, bufB, 336, nullptr, 0, nullptr, 0, tid);
    __syncthreads();
    mlp_layer<DE, 1>(bufB, 336, wE + DE * DE, eB + DE, nullptr, 0, bufY, 328, bufA, 336, tid);
    __syncthreads();
    ln_stats<DE>(bufY, 328, sMu, sRs, sPart, tid);
    // e_new out + build m_in = cat[x[col], e_new] into bufB
    for (int idx = tid; idx < 64 * DE; idx += 256) {
        int r = idx >> 8, c = idx & 255;
        float v = (bufY[r * 328 + c] - sMu[r]) * sRs[r] * eG[c] + eBt[c];
        long e = base + r;
        if (e < NEDGES) e_out[e * DE + c] = v;
        bufB[r * 336 + 64 + c] = (bf16)v;
        if (c < 64) bufB[r * 336 + c] = bufA[r * 336 + 64 + c];
    }
    __syncthreads();
    mlp_layer<DN1, 0>(bufB, 336, wN1, n1B, bufA, 336, nullptr, 0, nullptr, 0, tid);
    __syncthreads();
    mlp_layer<DN1, 1>(bufA, 336, wN1 + DN1 * DN1, n1B + DN1, nullptr, 0, bufY, 328, bufB, 336, tid);
    __syncthreads();
    ln_stats<DN1>(bufY, 328, sMu, sRs, sPart, tid);
    // scatter-mean numerator: atomic add message rows into agg[row]
    for (int idx = tid; idx < 64 * DN1; idx += 256) {
        int r = idx / DN1, c = idx - r * DN1;
        long e = base + r;
        if (e < NEDGES) {
            float v = (bufY[r * 328 + c] - sMu[r]) * sRs[r] * n1G[c] + n1Bt[c];
            atomAddF(&agg[(long)sRow[r] * DN1 + c], v);
        }
    }
}

// ---------------- node kernel ----------------------------------------------
__global__ __launch_bounds__(256) void node_kernel(
    const float* __restrict__ agg, const float* __restrict__ cnt,
    const float* __restrict__ u, const int* __restrict__ batch,
    const bf16* __restrict__ wN2, const float* __restrict__ n2B,
    const float* __restrict__ n2G, const float* __restrict__ n2Bt,
    float* __restrict__ x_out, float* __restrict__ gsum) {
    __shared__ bf16  bufA[64 * 400];
    __shared__ bf16  bufB[64 * 400];
    __shared__ float bufY[64 * 392];
    __shared__ int   sB[64];
    __shared__ float sMu[64], sRs[64], sInv[64];
    __shared__ float sPart[512];
    int tid = threadIdx.x;
    long base = (long)blockIdx.x * 64;

    if (tid < 64) {
        long n = base + tid; if (n >= NNODES) n = NNODES - 1;
        sB[tid] = batch[n];
        sInv[tid] = 1.0f / fmaxf(cnt[n], 1.0f);
    }
    __syncthreads();
    for (int idx = tid; idx < 64 * DN2; idx += 256) {
        int r = idx / DN2, c = idx - r * DN2;
        long n = base + r; if (n >= NNODES) n = NNODES - 1;
        float v = (c < DN1) ? agg[n * DN1 + c] * sInv[r] : u[sB[r] * 64 + (c - DN1)];
        bufA[r * 400 + c] = (bf16)v;
    }
    __syncthreads();
    mlp_layer<DN2, 0>(bufA, 400, wN2, n2B, bufB, 400, nullptr, 0, nullptr, 0, tid);
    __syncthreads();
    mlp_layer<DN2, 1>(bufB, 400, wN2 + DN2 * DN2, n2B + DN2, nullptr, 0, bufY, 392, bufA, 400, tid);
    __syncthreads();
    ln_stats<DN2>(bufY, 392, sMu, sRs, sPart, tid);
    for (int idx = tid; idx < 64 * DN2; idx += 256) {
        int r = idx / DN2, c = idx - r * DN2;
        float v = (bufY[r * 392 + c] - sMu[r]) * sRs[r] * n2G[c] + n2Bt[c];
        bufY[r * 392 + c] = v;
        long n = base + r;
        if (n < NNODES) x_out[n * DN2 + c] = v;
    }
    __syncthreads();
    // segmented per-graph sums (batch is sorted -> few segments per tile)
    for (int c = tid; c < DN2; c += 256) {
        float run = 0.f;
        int curb = sB[0];
        for (int r = 0; r < 64; ++r) {
            long n = base + r;
            if (n >= NNODES) break;
            int b = sB[r];
            if (b != curb) {
                if (run != 0.f) atomAddF(&gsum[curb * DN2 + c], run);
                run = 0.f; curb = b;
            }
            run += bufY[r * 392 + c];
        }
        if (run != 0.f) atomAddF(&gsum[curb * DN2 + c], run);
    }
}

// ---------------- global kernel (single workgroup, 64 graphs) ---------------
__global__ __launch_bounds__(256) void global_kernel(
    const float* __restrict__ u, const float* __restrict__ gsum,
    const float* __restrict__ gcnt, const bf16* __restrict__ wG,
    const float* __restrict__ gB, const float* __restrict__ gG,
    const float* __restrict__ gBt, float* __restrict__ u_out) {
    __shared__ bf16  bufA[64 * 464];
    __shared__ bf16  bufB[64 * 464];
    __shared__ float bufY[64 * 456];
    __shared__ float sMu[64], sRs[64];
    __shared__ float sPart[512];
    int tid = threadIdx.x;
    for (int idx = tid; idx < 64 * DG; idx += 256) {
        int r = idx / DG, c = idx - r * DG;
        float v = (c < 64) ? u[r * 64 + c]
                           : gsum[r * DN2 + (c - 64)] / fmaxf(gcnt[r], 1.0f);
        bufA[r * 464 + c] = (bf16)v;
    }
    __syncthreads();
    mlp_layer<DG, 0>(bufA, 464, wG, gB, bufB, 464, nullptr, 0, nullptr, 0, tid);
    __syncthreads();
    mlp_layer<DG, 1>(bufB, 464, wG + DG * DG, gB + DG, nullptr, 0, bufY, 456, bufA, 464, tid);
    __syncthreads();
    ln_stats<DG>(bufY, 456, sMu, sRs, sPart, tid);
    for (int idx = tid; idx < 64 * DG; idx += 256) {
        int r = idx / DG, c = idx - r * DG;
        u_out[r * DG + c] = (bufY[r * 456 + c] - sMu[r]) * sRs[r] * gG[c] + gBt[c];
    }
}

extern "C" void kernel_launch(void* const* d_in, const int* in_sizes, int n_in,
                              void* d_out, int out_size, void* d_ws, size_t ws_size,
                              hipStream_t stream) {
    (void)in_sizes; (void)n_in; (void)out_size; (void)ws_size;
    const float* x    = (const float*)d_in[0];
    const int*   ei   = (const int*)d_in[1];     // [2,E]: rows then cols
    const float* ea   = (const float*)d_in[2];
    const float* u    = (const float*)d_in[3];
    const int*   bat  = (const int*)d_in[4];
    const float* eW   = (const float*)d_in[5];
    const float* eB   = (const float*)d_in[6];
    const float* eG   = (const float*)d_in[7];
    const float* eBt  = (const float*)d_in[8];
    const float* n1W  = (const float*)d_in[9];
    const float* n1B  = (const float*)d_in[10];
    const float* n1G  = (const float*)d_in[11];
    const float* n1Bt = (const float*)d_in[12];
    const float* n2W  = (const float*)d_in[13];
    const float* n2B  = (const float*)d_in[14];
    const float* n2G  = (const float*)d_in[15];
    const float* n2Bt = (const float*)d_in[16];
    const float* gW   = (const float*)d_in[17];
    const float* gB   = (const float*)d_in[18];
    const float* gG   = (const float*)d_in[19];
    const float* gBt  = (const float*)d_in[20];

    char* ws = (char*)d_ws;
    bf16*  wE   = (bf16*)(ws + WS_WE);
    bf16*  wN1  = (bf16*)(ws + WS_WN1);
    bf16*  wN2  = (bf16*)(ws + WS_WN2);
    bf16*  wG   = (bf16*)(ws + WS_WG);
    float* agg  = (float*)(ws + WS_AGG);
    float* cnt  = (float*)(ws + WS_CNT);
    float* gsum = (float*)(ws + WS_GSUM);
    float* gcnt = (float*)(ws + WS_GCNT);

    float* x_out = (float*)d_out;
    float* e_out = x_out + (size_t)NNODES * DN2;
    float* u_out = e_out + (size_t)NEDGES * DE;

    prep_weights<<<(2 * DE * DE + 255) / 256, 256, 0, stream>>>(eW, wE, DE);
    prep_weights<<<(2 * DN1 * DN1 + 255) / 256, 256, 0, stream>>>(n1W, wN1, DN1);
    prep_weights<<<(2 * DN2 * DN2 + 255) / 256, 256, 0, stream>>>(n2W, wN2, DN2);
    prep_weights<<<(2 * DG * DG + 255) / 256, 256, 0, stream>>>(gW, wG, DG);
    zero_ws<<<2048, 256, 0, stream>>>((float4*)(ws + WS_AGG), WS_ZERO_BYTES / 16);
    counts_kernel<<<(NEDGES + NNODES + 255) / 256, 256, 0, stream>>>(ei, bat, cnt, gcnt);
    edge_kernel<<<(NEDGES + 63) / 64, 256, 0, stream>>>(
        x, ei, ei + NEDGES, ea, u, bat, wE, eB, eG, eBt, wN1, n1B, n1G, n1Bt, e_out, agg);
    node_kernel<<<(NNODES + 63) / 64, 256, 0, stream>>>(
        agg, cnt, u, bat, wN2, n2B, n2G, n2Bt, x_out, gsum);
    global_kernel<<<1, 256, 0, stream>>>(u, gsum, gcnt, wG, gB, gG, gBt, u_out);
}
